// EnhancedAttention_27839978012915
// MI455X (gfx1250) — compile-verified, matrix operand order checked by reading, not run on hardware
//
#include <hip/hip_runtime.h>

// ---------------------------------------------------------------------------
// EnhancedAttention for MI455X (gfx1250, wave32, WMMA + async-to-LDS)
//   B=8, L=4096, D_MODEL=1024, H=16, Dh=64
//   stage 0: w_qkv / w_out  -> transposed bf16 [N][K] (one-time, 17 MB)
//   stage 1: qkv = x @ w_qkv + b_qkv         (bf16 WMMA, f32 accum, async DMA)
//   stage 2: per-token 16x16 head attention  (f32 WMMA 16x16x4)
//   stage 3: out = attn_out @ w_out + b_out  (bf16 WMMA, f32 accum, async DMA)
// ---------------------------------------------------------------------------

typedef __bf16 bf16x8  __attribute__((ext_vector_type(8)));
typedef __bf16 bf16x16 __attribute__((ext_vector_type(16)));
typedef float  f32x2   __attribute__((ext_vector_type(2)));
typedef float  f32x4   __attribute__((ext_vector_type(4)));
typedef float  f32x8   __attribute__((ext_vector_type(8)));

#define BQ     8
#define LQ     4096
#define DMODEL 1024
#define NHEAD  16
#define HDIM   64
#define NQKV   3072              // 3*H*Dh
#define NTOK   (BQ * LQ)         // 32768

// ---------------------------------------------------------------------------
// CDNA5 async copy: global -> LDS, ASYNCcnt-tracked, GVS addressing
// (SGPR 64-bit base + per-lane 32-bit VGPR offset).
// ---------------------------------------------------------------------------
__device__ __forceinline__ void async_copy_b128(unsigned lds_off,
                                                unsigned long long gbase,
                                                unsigned goff) {
    asm volatile("global_load_async_to_lds_b128 %0, %1, %2"
                 :: "v"(lds_off), "v"(goff), "s"(gbase) : "memory");
}
__device__ __forceinline__ void wait_async0() {
    asm volatile("s_wait_asynccnt 0x0" ::: "memory");
}
__device__ __forceinline__ unsigned lds_addr(const void* p) {
    // flat shared pointer: low 32 bits == wave-relative LDS byte offset
    return (unsigned)(unsigned long long)(uintptr_t)p;
}

// ---------------------------------------------------------------------------
// stage 0: Wt[n][k] = (bf16) W[k][n]   (32x32 LDS tile transpose)
// ---------------------------------------------------------------------------
__global__ __launch_bounds__(256) void transpose_convert_bf16(
    const float* __restrict__ W, __bf16* __restrict__ Wt, int K, int N)
{
    __shared__ float t[32][33];
    const int k0 = blockIdx.x * 32, n0 = blockIdx.y * 32;
    const int tx = threadIdx.x & 31, ty = threadIdx.x >> 5;   // ty 0..7
#pragma unroll
    for (int i = ty; i < 32; i += 8)
        t[i][tx] = W[(size_t)(k0 + i) * N + (n0 + tx)];
    __syncthreads();
#pragma unroll
    for (int i = ty; i < 32; i += 8)
        Wt[(size_t)(n0 + i) * K + (k0 + tx)] = (__bf16)t[tx][i];
}

// ---------------------------------------------------------------------------
// GEMM: C[M,N] = A[M,K](fp32) @ Wt[N,K](bf16) + bias[N]
// Block: 256 threads = 8 waves. Tile 128(M) x 128(N), K-step 32.
// Double-buffered LDS filled by global_load_async_to_lds_b128.
//   As: fp32 [128][36]  (row stride 144B: 16B-aligned, conflict-free frags)
//   Bs: bf16 [128][40]  (row stride  80B: 16B-aligned, conflict-free frags)
// Wave w owns M-rows [w*16, w*16+16), all 8 N-tiles: 8 x v_wmma per K-step.
// ---------------------------------------------------------------------------
#define BM 128
#define BN 128
#define BK 32
#define APAD 36
#define BPAD 40

__global__ __launch_bounds__(256) void gemm_bias_wmma(
    const float* __restrict__ A, const __bf16* __restrict__ Wt,
    const float* __restrict__ bias, float* __restrict__ C,
    int M, int N, int K)
{
    __shared__ float  As[2][BM][APAD];   // 36 KB
    __shared__ __bf16 Bs[2][BM][BPAD];   // 20 KB

    const int tid  = threadIdx.x;
    const int m0   = blockIdx.y * BM;
    const int n0   = blockIdx.x * BN;
    const int wv   = tid >> 5;
    const int lane = tid & 31;
    const int half = lane >> 4;
    const int mr   = lane & 15;
    const int ks   = half * 8;           // K sub-chunk base per ISA A/B layout

    const unsigned long long Abase = (unsigned long long)(uintptr_t)A;
    const unsigned long long Bbase = (unsigned long long)(uintptr_t)Wt;

    // per-thread async-copy chunk coordinates (fixed across K-steps)
    int arow[4], ac4[4];
#pragma unroll
    for (int i = 0; i < 4; ++i) {                 // A: 1024 float4 chunks
        int f4 = i * 256 + tid;
        arow[i] = f4 >> 3;                        // 8 float4 per row
        ac4[i]  = f4 & 7;
    }
    int brow[2], bkc[2];
#pragma unroll
    for (int i = 0; i < 2; ++i) {                 // B: 512 bf16x8 chunks
        int ch = i * 256 + tid;
        brow[i] = ch >> 2;                        // 4 chunks per row
        bkc[i]  = ch & 3;
    }

    // issue async fills for one K-tile into buffer b
    auto issue_stage = [&](int b, int k0) {
#pragma unroll
        for (int i = 0; i < 4; ++i) {
            unsigned go = (unsigned)(((m0 + arow[i]) * K + k0 + ac4[i] * 4) *
                                     (int)sizeof(float));
            async_copy_b128(lds_addr(&As[b][arow[i]][ac4[i] * 4]), Abase, go);
        }
#pragma unroll
        for (int i = 0; i < 2; ++i) {
            unsigned go = (unsigned)(((n0 + brow[i]) * K + k0 + bkc[i] * 8) *
                                     (int)sizeof(__bf16));
            async_copy_b128(lds_addr(&Bs[b][brow[i]][bkc[i] * 8]), Bbase, go);
        }
    };

    f32x8 acc[8];
#pragma unroll
    for (int j = 0; j < 8; ++j)
        acc[j] = (f32x8){0.f,0.f,0.f,0.f,0.f,0.f,0.f,0.f};

    issue_stage(0, 0);
    int buf = 0;
    for (int k0 = 0; k0 < K; k0 += BK) {
        wait_async0();          // our fills for buf are done
        __syncthreads();        // everyone's fills done; prior reads retired
        if (k0 + BK < K) issue_stage(buf ^ 1, k0 + BK);

        // A fragment: lane = row (wv*16+mr), K = {ks..ks+7, ks+16..ks+23}
        const float* ar = &As[buf][wv * 16 + mr][0];
        f32x4 a0 = *(const f32x4*)(ar + ks);
        f32x4 a1 = *(const f32x4*)(ar + ks + 4);
        f32x4 a2 = *(const f32x4*)(ar + ks + 16);
        f32x4 a3 = *(const f32x4*)(ar + ks + 20);
        union { bf16x16 v; __bf16 e[16]; } fa;
#pragma unroll
        for (int t = 0; t < 4; ++t) {
            fa.e[t]      = (__bf16)a0[t];
            fa.e[4 + t]  = (__bf16)a1[t];
            fa.e[8 + t]  = (__bf16)a2[t];
            fa.e[12 + t] = (__bf16)a3[t];
        }

#pragma unroll
        for (int j = 0; j < 8; ++j) {
            // B fragment: lane = column (j*16+mr), same K chunks, already bf16
            const __bf16* br = &Bs[buf][j * 16 + mr][0];
            union { bf16x16 v; bf16x8 h[2]; } fb;
            fb.h[0] = *(const bf16x8*)(br + ks);
            fb.h[1] = *(const bf16x8*)(br + ks + 16);
            acc[j] = __builtin_amdgcn_wmma_f32_16x16x32_bf16(
                false, fa.v, false, fb.v, (short)0, acc[j], false, false);
        }
        buf ^= 1;
    }

    // epilogue: C layout -> lane owns column mr, rows r + 8*half
#pragma unroll
    for (int j = 0; j < 8; ++j) {
        int n = n0 + j * 16 + mr;
        float bv = bias[n];
#pragma unroll
        for (int r = 0; r < 8; ++r) {
            int m = m0 + wv * 16 + r + half * 8;
            C[(size_t)m * N + n] = acc[j][r] + bv;
        }
    }
}

// ---------------------------------------------------------------------------
// per-token head attention (1 wave = 1 token)
// qkv row per token: head h at [h*192 + {0..63 q, 64..127 k, 128..191 v}]
// S = Q(16x64) @ K'^T(64x16) / 8 ; P = softmax_rows(S) ; O = P @ V(16x64)
// ---------------------------------------------------------------------------
__global__ __launch_bounds__(128) void attn_heads_wmma(
    const float* __restrict__ qkv, float* __restrict__ attn_out)
{
    __shared__ float Ps[4][16][17];

    const int tid   = threadIdx.x;
    const int wv    = tid >> 5;
    const int lane  = tid & 31;
    const int half  = lane >> 4;
    const int mr    = lane & 15;
    const int token = blockIdx.x * 4 + wv;
    const int l     = token % LQ;
    const float* tok = qkv + (size_t)token * NQKV;

    // ---- S = Q @ (K + pos_enc)^T, f32 WMMA 16x16x4, 16 K-steps ----
    f32x8 s = (f32x8){0.f,0.f,0.f,0.f,0.f,0.f,0.f,0.f};
    const float cpe = 0.14391156f;    // ln(10000)/64
#pragma unroll
    for (int kb = 0; kb < HDIM; kb += 4) {
        int kk = kb + half * 2;       // even
        f32x2 a = { tok[mr * 192 + kk], tok[mr * 192 + kk + 1] };
        float sn, cs;
        __sincosf((float)l * __expf(-(float)kk * cpe), &sn, &cs);
        f32x2 b = { tok[mr * 192 + 64 + kk]     + sn,
                    tok[mr * 192 + 64 + kk + 1] + cs };
        s = __builtin_amdgcn_wmma_f32_16x16x4_f32(
            false, a, false, b, (short)0, s, false, false);
    }

    // ---- softmax over columns (N across lanes within each 16-lane half) ----
    f32x8 p;
#pragma unroll
    for (int r = 0; r < 8; ++r) {
        float v = s[r] * 0.125f;      // / sqrt(64)
        float mx = v;
#pragma unroll
        for (int off = 8; off > 0; off >>= 1)
            mx = fmaxf(mx, __shfl_xor(mx, off, 16));
        float e = __expf(v - mx);
        float sum = e;
#pragma unroll
        for (int off = 8; off > 0; off >>= 1)
            sum += __shfl_xor(sum, off, 16);
        p[r] = e / sum;
    }

    // reshape P from C-layout to A-layout via LDS
#pragma unroll
    for (int r = 0; r < 8; ++r)
        Ps[wv][r + half * 8][mr] = p[r];
    __syncthreads();

    // ---- O = P(16x16) @ V(16x64), f32 WMMA, 4 K-steps, 4 N-tiles ----
    f32x8 o[4];
#pragma unroll
    for (int j = 0; j < 4; ++j)
        o[j] = (f32x8){0.f,0.f,0.f,0.f,0.f,0.f,0.f,0.f};
#pragma unroll
    for (int kb = 0; kb < 16; kb += 4) {
        int kk = kb + half * 2;
        f32x2 a = { Ps[wv][mr][kk], Ps[wv][mr][kk + 1] };
#pragma unroll
        for (int j = 0; j < 4; ++j) {
            int d = j * 16 + mr;
            f32x2 b = { tok[kk * 192 + 128 + d],
                        tok[(kk + 1) * 192 + 128 + d] };
            o[j] = __builtin_amdgcn_wmma_f32_16x16x4_f32(
                false, a, false, b, (short)0, o[j], false, false);
        }
    }

    float* op = attn_out + (size_t)token * (NHEAD * HDIM);
#pragma unroll
    for (int j = 0; j < 4; ++j)
#pragma unroll
        for (int r = 0; r < 8; ++r)
            op[(r + half * 8) * HDIM + (j * 16 + mr)] = o[j][r];
}

// ---------------------------------------------------------------------------
extern "C" void kernel_launch(void* const* d_in, const int* in_sizes, int n_in,
                              void* d_out, int out_size, void* d_ws, size_t ws_size,
                              hipStream_t stream) {
    const float* x     = (const float*)d_in[0];
    const float* w_qkv = (const float*)d_in[1];
    const float* b_qkv = (const float*)d_in[2];
    const float* w_out = (const float*)d_in[3];
    const float* b_out = (const float*)d_in[4];
    float* out = (float*)d_out;

    char* ws = (char*)d_ws;
    float*  qkv      = (float*)ws;                                 // 402.6 MB
    float*  attn_out = (float*)(ws + (size_t)NTOK * NQKV * 4);     // +134.2 MB
    __bf16* wt_qkv   = (__bf16*)(ws + (size_t)NTOK * NQKV * 4
                                    + (size_t)NTOK * DMODEL * 4);  // +6.3 MB
    __bf16* wt_out   = (__bf16*)((char*)wt_qkv
                                    + (size_t)NQKV * DMODEL * 2);  // +2.1 MB

    // stage 0: one-time weight transpose + bf16 convert
    transpose_convert_bf16<<<dim3(DMODEL / 32, NQKV / 32), 256, 0, stream>>>(
        w_qkv, wt_qkv, DMODEL, NQKV);
    transpose_convert_bf16<<<dim3(DMODEL / 32, DMODEL / 32), 256, 0, stream>>>(
        w_out, wt_out, DMODEL, DMODEL);

    // stage 1: qkv = x @ w_qkv + b_qkv   [32768,1024]x[1024,3072]
    gemm_bias_wmma<<<dim3(NQKV / BN, NTOK / BM), 256, 0, stream>>>(
        x, wt_qkv, b_qkv, qkv, NTOK, NQKV, DMODEL);

    // stage 2: per-token head attention (4 tokens per 128-thread block)
    attn_heads_wmma<<<NTOK / 4, 128, 0, stream>>>(qkv, attn_out);

    // stage 3: out = attn_out @ w_out + b_out  [32768,1024]x[1024,1024]
    gemm_bias_wmma<<<dim3(DMODEL / BN, NTOK / BM), 256, 0, stream>>>(
        attn_out, wt_out, b_out, out, NTOK, DMODEL, DMODEL);
}
